// AttentionMoudle_63428077027828
// MI455X (gfx1250) — matrix operand order repaired and checked by reading, not compiled, not run on hardware
//
#include <hip/hip_runtime.h>

typedef __attribute__((ext_vector_type(16))) _Float16 v16h;
typedef __attribute__((ext_vector_type(8)))  float    v8f;
typedef __attribute__((ext_vector_type(2)))  _Float16 h2;
typedef __attribute__((ext_vector_type(2)))  int      v2i;

#define LDSK 36   // f16 row stride in LDS: 72B -> 8B-aligned, conflict-free across 16 rows

#ifndef __has_builtin
#define __has_builtin(x) 0
#endif
#if __has_builtin(__builtin_amdgcn_global_load_async_to_lds_b64)
#define HAS_ASYNC_LDS 1
#else
#define HAS_ASYNC_LDS 0
#endif

// 8-byte global->LDS copy: async (ASYNCcnt-tracked, bypasses VGPRs) when available.
__device__ __forceinline__ void copy8_g2l(_Float16* l, const _Float16* g) {
#if HAS_ASYNC_LDS
  typedef v2i __attribute__((address_space(1)))* gp_t;
  typedef v2i __attribute__((address_space(3)))* lp_t;
  __builtin_amdgcn_global_load_async_to_lds_b64((gp_t)g, (lp_t)l, 0, 0);
#else
  *(uint2*)l = *(const uint2*)g;
#endif
}

__device__ __forceinline__ void async_fence() {
#if HAS_ASYNC_LDS
#if __has_builtin(__builtin_amdgcn_s_wait_asynccnt)
  __builtin_amdgcn_s_wait_asynccnt(0);
#else
  asm volatile("s_wait_asynccnt 0x0" ::: "memory");
#endif
#endif
}

// ---- WMMA fragment loaders (ISA 7.12.2 layouts), LDS tiles are [rows][32 K] stride LDSK ----

// A 16x32 f16: lane m=lane&15, g=lane>>4; VGPR v half h -> K = 16*(v>=4) + 8*g + 2*(v&3) + h
__device__ __forceinline__ v16h frag_A(const _Float16* t, int lane) {
  const int m = lane & 15, g = lane >> 4;
  v16h a;
#pragma unroll
  for (int v = 0; v < 8; ++v) {
    const int k = ((v & 4) << 2) + (g << 3) + ((v & 3) << 1);
    h2 p = *(const h2*)(t + m * LDSK + k);
    a[2 * v] = p.x;
    a[2 * v + 1] = p.y;
  }
  return a;
}

// B 32x16 f16 staged as [n][k]: lane n=lane&15, g=lane>>4; VGPR v half h -> K = 16*(v>=4) + 8*g + 2*(v&3) + h
__device__ __forceinline__ v16h frag_B(const _Float16* t, int lane) {
  const int n = lane & 15, g = lane >> 4;
  v16h b;
#pragma unroll
  for (int v = 0; v < 8; ++v) {
    const int k = ((v & 4) << 2) + (g << 3) + ((v & 3) << 1);
    h2 p = *(const h2*)(t + n * LDSK + k);
    b[2 * v] = p.x;
    b[2 * v + 1] = p.y;
  }
  return b;
}

__device__ __forceinline__ v8f wmma16(v16h a, v16h b, v8f c) {
  return __builtin_amdgcn_wmma_f32_16x16x32_f16(false, a, false, b, (short)0, c,
                                                false, false);
}

// Block tile 128(M) x 128(N), 8 waves as 4(M) x 2(N); wave tile 32x64 = 2x4 WMMA tiles.
#define WAVE_COMPUTE()                                                          \
  const v16h a0 = frag_A(As + (wm * 32) * LDSK, lane);                          \
  const v16h a1 = frag_A(As + (wm * 32 + 16) * LDSK, lane);                     \
  const v16h b0 = frag_B(Bs + (wn * 64) * LDSK, lane);                          \
  const v16h b1 = frag_B(Bs + (wn * 64 + 16) * LDSK, lane);                     \
  const v16h b2 = frag_B(Bs + (wn * 64 + 32) * LDSK, lane);                     \
  const v16h b3 = frag_B(Bs + (wn * 64 + 48) * LDSK, lane);                     \
  acc[0][0] = wmma16(a0, b0, acc[0][0]);                                        \
  acc[0][1] = wmma16(a0, b1, acc[0][1]);                                        \
  acc[0][2] = wmma16(a0, b2, acc[0][2]);                                        \
  acc[0][3] = wmma16(a0, b3, acc[0][3]);                                        \
  acc[1][0] = wmma16(a1, b0, acc[1][0]);                                        \
  acc[1][1] = wmma16(a1, b1, acc[1][1]);                                        \
  acc[1][2] = wmma16(a1, b2, acc[1][2]);                                        \
  acc[1][3] = wmma16(a1, b3, acc[1][3]);

// =====================================================================================
// Projection: P[m,e] = sum_d X[m,d] * W[e,d]   (X: [16384,512] f32, W: [512,512] f32)
// grid: 512 blocks -> mt = bx>>2 (0..127), et = bx&3 (0..3). Output f16.
// =====================================================================================
__global__ __launch_bounds__(256) void proj_kernel(const float* __restrict__ X,
                                                   const float* __restrict__ W,
                                                   _Float16* __restrict__ P) {
  __shared__ _Float16 As[128 * LDSK];
  __shared__ _Float16 Bs[128 * LDSK];
  const int tid = threadIdx.x;
  const int lane = tid & 31, wid = tid >> 5;
  const int wm = wid >> 1, wn = wid & 1;
  const int mt = blockIdx.x >> 2, et = blockIdx.x & 3;
  const int row0 = mt * 128;
  const int e0 = et * 128;

  v8f acc[2][4] = {};
  for (int k0 = 0; k0 < 512; k0 += 32) {
    __syncthreads();
#pragma unroll
    for (int it = 0; it < 4; ++it) {  // A: 128x32 f32 -> f16
      const int idx = (it * 256 + tid) * 4;
      const int r = idx >> 5, c = idx & 31;
      const float4 f = *(const float4*)(X + (size_t)(row0 + r) * 512 + k0 + c);
      _Float16* d = As + r * LDSK + c;
      h2 lo, hi;
      lo.x = (_Float16)f.x; lo.y = (_Float16)f.y;
      hi.x = (_Float16)f.z; hi.y = (_Float16)f.w;
      *(h2*)(d) = lo;
      *(h2*)(d + 2) = hi;
    }
#pragma unroll
    for (int it = 0; it < 4; ++it) {  // B: 128x32 = W rows e0..e0+127 ([e][d] row-major)
      const int idx = (it * 256 + tid) * 4;
      const int r = idx >> 5, c = idx & 31;
      const float4 f = *(const float4*)(W + (size_t)(e0 + r) * 512 + k0 + c);
      _Float16* d = Bs + r * LDSK + c;
      h2 lo, hi;
      lo.x = (_Float16)f.x; lo.y = (_Float16)f.y;
      hi.x = (_Float16)f.z; hi.y = (_Float16)f.w;
      *(h2*)(d) = lo;
      *(h2*)(d + 2) = hi;
    }
    __syncthreads();
    WAVE_COMPUTE()
  }
  const int nlane = lane & 15, mg = (lane >> 4) << 3;
#pragma unroll
  for (int ti = 0; ti < 2; ++ti)
#pragma unroll
    for (int tj = 0; tj < 4; ++tj)
#pragma unroll
      for (int r = 0; r < 8; ++r) {
        const size_t m = (size_t)(row0 + wm * 32 + ti * 16 + mg + r);
        const int e = e0 + wn * 64 + tj * 16 + nlane;
        P[m * 512 + e] = (_Float16)acc[ti][tj][r];
      }
}

// =====================================================================================
// Scores: S[b,q,k] = sum_d qp[b,q,d] * kp[b,k,d]  -> f32 (pre-softmax)
// Both operands f16, staged via async global->LDS copies (ASYNCcnt path).
// grid: 2048 -> b = bx>>8, mt = (bx>>4)&15, nt = bx&15
// =====================================================================================
__global__ __launch_bounds__(256) void scores_kernel(const _Float16* __restrict__ Q,
                                                     const _Float16* __restrict__ K,
                                                     float* __restrict__ S) {
  __shared__ _Float16 As[128 * LDSK];
  __shared__ _Float16 Bs[128 * LDSK];
  const int tid = threadIdx.x;
  const int lane = tid & 31, wid = tid >> 5;
  const int wm = wid >> 1, wn = wid & 1;
  const int b = blockIdx.x >> 8;
  const int rem = blockIdx.x & 255;
  const int mt = rem >> 4, nt = rem & 15;
  const _Float16* Qb = Q + (size_t)b * 2048 * 512 + (size_t)mt * 128 * 512;
  const _Float16* Kb = K + (size_t)b * 2048 * 512 + (size_t)nt * 128 * 512;
  float* Sb = S + (size_t)b * 2048 * 2048 + (size_t)mt * 128 * 2048;

  v8f acc[2][4] = {};
  for (int k0 = 0; k0 < 512; k0 += 32) {
    __syncthreads();
#pragma unroll
    for (int it = 0; it < 4; ++it) {  // A: 128x32 f16
      const int idx = (it * 256 + tid) * 4;
      const int r = idx >> 5, c = idx & 31;
      copy8_g2l(As + r * LDSK + c, Qb + (size_t)r * 512 + k0 + c);
    }
#pragma unroll
    for (int it = 0; it < 4; ++it) {  // B: 128x32 f16 (rows of kp == rows of B^T)
      const int idx = (it * 256 + tid) * 4;
      const int r = idx >> 5, c = idx & 31;
      copy8_g2l(Bs + r * LDSK + c, Kb + (size_t)r * 512 + k0 + c);
    }
    async_fence();
    __syncthreads();
    WAVE_COMPUTE()
  }
  const int nlane = lane & 15, mg = (lane >> 4) << 3;
#pragma unroll
  for (int ti = 0; ti < 2; ++ti)
#pragma unroll
    for (int tj = 0; tj < 4; ++tj)
#pragma unroll
      for (int r = 0; r < 8; ++r) {
        const size_t m = (size_t)(wm * 32 + ti * 16 + mg + r);
        const int n = nt * 128 + wn * 64 + tj * 16 + nlane;
        Sb[m * 2048 + n] = acc[ti][tj][r];
      }
}

// =====================================================================================
// Row softmax in place on [16384 rows x 2048]: one 256-thread block per row.
// =====================================================================================
__global__ __launch_bounds__(256) void softmax_kernel(float* __restrict__ S) {
  __shared__ float red[256];
  float* p = S + (size_t)blockIdx.x * 2048;
  const int tid = threadIdx.x;
  float x[8];
  float m = -3.4e38f;
#pragma unroll
  for (int i = 0; i < 8; ++i) {
    x[i] = p[tid + i * 256];
    m = fmaxf(m, x[i]);
  }
  red[tid] = m;
  __syncthreads();
  for (int s = 128; s > 0; s >>= 1) {
    if (tid < s) red[tid] = fmaxf(red[tid], red[tid + s]);
    __syncthreads();
  }
  m = red[0];
  __syncthreads();
  float sum = 0.f;
#pragma unroll
  for (int i = 0; i < 8; ++i) {
    x[i] = __expf(x[i] - m);
    sum += x[i];
  }
  red[tid] = sum;
  __syncthreads();
  for (int s = 128; s > 0; s >>= 1) {
    if (tid < s) red[tid] += red[tid + s];
    __syncthreads();
  }
  const float inv = 1.0f / red[0];
#pragma unroll
  for (int i = 0; i < 8; ++i) p[tid + i * 256] = x[i] * inv;
}

// =====================================================================================
// Output: O[b,q,d] = sum_k (scale*P[b,q,k]) * vp[b,k,d]
// A = P (f32, scale folded into cvt), B = vp staged transposed [n][k] into LDS.
// grid: 512 -> b = bx>>6, mt = (bx>>2)&15, nt = bx&3. K = 2048 in steps of 32.
// =====================================================================================
__global__ __launch_bounds__(256) void outgemm_kernel(const float* __restrict__ P,
                                                      const _Float16* __restrict__ V,
                                                      float* __restrict__ O,
                                                      float scale) {
  __shared__ _Float16 As[128 * LDSK];
  __shared__ _Float16 Bs[128 * LDSK];
  const int tid = threadIdx.x;
  const int lane = tid & 31, wid = tid >> 5;
  const int wm = wid >> 1, wn = wid & 1;
  const int b = blockIdx.x >> 6;
  const int rem = blockIdx.x & 63;
  const int mt = rem >> 2, nt = rem & 3;
  const int n0 = nt * 128;
  const float* Pb = P + (size_t)b * 2048 * 2048 + (size_t)mt * 128 * 2048;
  const _Float16* Vb = V + (size_t)b * 2048 * 512;
  float* Ob = O + (size_t)b * 2048 * 512 + (size_t)mt * 128 * 512;

  v8f acc[2][4] = {};
  for (int k0 = 0; k0 < 2048; k0 += 32) {
    __syncthreads();
#pragma unroll
    for (int it = 0; it < 4; ++it) {  // A: 128x32 f32 probs, scale + cvt
      const int idx = (it * 256 + tid) * 4;
      const int r = idx >> 5, c = idx & 31;
      const float4 f = *(const float4*)(Pb + (size_t)r * 2048 + k0 + c);
      _Float16* d = As + r * LDSK + c;
      h2 lo, hi;
      lo.x = (_Float16)(f.x * scale); lo.y = (_Float16)(f.y * scale);
      hi.x = (_Float16)(f.z * scale); hi.y = (_Float16)(f.w * scale);
      *(h2*)(d) = lo;
      *(h2*)(d + 2) = hi;
    }
#pragma unroll
    for (int it = 0; it < 8; ++it) {  // B: vp tile 32(k) x 128(n) -> transposed Bs[n][k]
      const int idx = (it * 256 + tid) * 2;
      const int k = idx >> 7, n = idx & 127;
      h2 p = *(const h2*)(Vb + (size_t)(k0 + k) * 512 + n0 + n);
      Bs[n * LDSK + k] = p.x;
      Bs[(n + 1) * LDSK + k] = p.y;
    }
    __syncthreads();
    WAVE_COMPUTE()
  }
  const int nlane = lane & 15, mg = (lane >> 4) << 3;
#pragma unroll
  for (int ti = 0; ti < 2; ++ti)
#pragma unroll
    for (int tj = 0; tj < 4; ++tj)
#pragma unroll
      for (int r = 0; r < 8; ++r) {
        const size_t m = (size_t)(wm * 32 + ti * 16 + mg + r);
        const int n = n0 + wn * 64 + tj * 16 + nlane;
        Ob[m * 512 + n] = acc[ti][tj][r];
      }
}

extern "C" void kernel_launch(void* const* d_in, const int* in_sizes, int n_in,
                              void* d_out, int out_size, void* d_ws, size_t ws_size,
                              hipStream_t stream) {
  (void)in_sizes; (void)n_in; (void)out_size; (void)ws_size;
  const float* q  = (const float*)d_in[0];
  const float* k  = (const float*)d_in[1];
  const float* v  = (const float*)d_in[2];
  const float* Wq = (const float*)d_in[3];
  const float* Wk = (const float*)d_in[4];
  const float* Wv = (const float*)d_in[5];

  float* attn = (float*)d_out;                            // [8,2048,2048] softmax probs
  float* obuf = attn + (size_t)8 * 2048 * 2048;           // [8,2048,512]

  _Float16* qp = (_Float16*)d_ws;                         // 3 x 16 MB f16 projections
  _Float16* kp = qp + (size_t)8 * 2048 * 512;
  _Float16* vp = kp + (size_t)8 * 2048 * 512;

  const dim3 blk(256);
  proj_kernel<<<512, blk, 0, stream>>>(q, Wq, qp);
  proj_kernel<<<512, blk, 0, stream>>>(k, Wk, kp);
  proj_kernel<<<512, blk, 0, stream>>>(v, Wv, vp);
  scores_kernel<<<2048, blk, 0, stream>>>(qp, kp, attn);
  softmax_kernel<<<16384, blk, 0, stream>>>(attn);
  const float scale = 0.044194173824159216f;              // 512^-0.5
  outgemm_kernel<<<512, blk, 0, stream>>>(attn, vp, obuf, scale);
}
